// MnistPrllSplineKAN_22771916603482
// MI455X (gfx1250) — matrix operand order repaired and checked by reading, not compiled, not run on hardware
//
#include <hip/hip_runtime.h>
#include <stdint.h>

typedef __attribute__((ext_vector_type(16))) _Float16 v16h;
typedef __attribute__((ext_vector_type(8)))  _Float16 v8h;
typedef __attribute__((ext_vector_type(8)))  float    v8f;
typedef __attribute__((ext_vector_type(4)))  unsigned int v4u;
typedef __attribute__((ext_vector_type(4)))  int      v4i;
typedef __attribute__((ext_vector_type(8)))  int      v8i;

#define IN_DIM   784
#define HEADS    10
#define OUT_DIM  16
#define NGRID    12
#define K_SP     6272          // 784 * 8 spline features
#define K_TOT    7072          // 6272 + 800 (784 silu features padded to 800)
#define NCOL     160           // HEADS * OUT_DIM
#define MT       64            // batch rows per workgroup
#define BT       128           // 4 waves of 32
#define A_STR    40            // halves per As row (80 B: 16B-aligned, non-pow2 banks)
#define B_STR    40            // halves per Bs row (64 B data + 16 B pad)
#define CH_SP    196           // 6272 / 32
#define CH_TOT   221           // + 800 / 32
#define YB_STR   161           // floats per ybuf row (conflict-breaking pad)

#define AS_BYTES (MT * A_STR * 2)          //  5120
#define BS_BYTES (NCOL * B_STR * 2)        // 12800 per buffer
#define STAGE_BYTES (AS_BYTES + 2 * BS_BYTES)
#define YB_BYTES (MT * YB_STR * 4)         // 41216
#define SMEM_BYTES (YB_BYTES > STAGE_BYTES ? YB_BYTES : STAGE_BYTES)

#if defined(__has_builtin)
# if __has_builtin(__builtin_amdgcn_tensor_load_to_lds) && __has_builtin(__builtin_amdgcn_s_wait_tensorcnt)
#  define HAVE_TDM 1
# endif
#endif
#ifndef HAVE_TDM
# define HAVE_TDM 0
#endif

// ---------------------------------------------------------------------------
// Stage 0: fuse coef*scale_sp*mask and scale_base*mask into one f16 weight
// matrix W2[n][k], n = h*16+o, k in [0,7072). Pad region k>=7056 is zero.
// ---------------------------------------------------------------------------
__global__ __launch_bounds__(256) void kan_prep_w(
    const float* __restrict__ coef, const float* __restrict__ scale_base,
    const float* __restrict__ scale_sp, const float* __restrict__ mask,
    _Float16* __restrict__ W2)
{
  int idx = blockIdx.x * 256 + threadIdx.x;
  if (idx >= NCOL * K_TOT) return;
  int n = idx / K_TOT;
  int k = idx - n * K_TOT;
  int h = n >> 4, o = n & 15;
  float v = 0.0f;
  if (k < K_SP) {
    int d = k >> 3, g = k & 7;
    int hd = (h * IN_DIM + d) * OUT_DIM + o;
    v = coef[hd * 8 + g] * scale_sp[hd] * mask[hd];
  } else if (k < K_SP + IN_DIM) {
    int d = k - K_SP;
    int hd = (h * IN_DIM + d) * OUT_DIM + o;
    v = scale_base[hd] * mask[hd];
  }
  W2[idx] = (_Float16)v;
}

// ---------------------------------------------------------------------------
// Cox-de Boor order-3 B-spline: 8 basis values on the shared 12-knot grid.
// ---------------------------------------------------------------------------
__device__ __forceinline__ v8h spline8(float x, const float* __restrict__ g) {
  float B[11];
#pragma unroll
  for (int i = 0; i < 11; ++i)
    B[i] = (x >= g[i] && x < g[i + 1]) ? 1.0f : 0.0f;
#pragma unroll
  for (int p = 1; p <= 3; ++p) {
#pragma unroll
    for (int i = 0; i + p < 11; ++i) {
      float l = (x - g[i])         / (g[i + p]     - g[i]);
      float r = (g[i + p + 1] - x) / (g[i + p + 1] - g[i + 1]);
      B[i] = l * B[i] + r * B[i + 1];
    }
  }
  v8h o;
#pragma unroll
  for (int i = 0; i < 8; ++i) o[i] = (_Float16)B[i];
  return o;
}

__device__ __forceinline__ float silu(float x) {
  return x / (1.0f + __expf(-x));
}

#if HAVE_TDM
// ---------------------------------------------------------------------------
// TDM: DMA one 32K x 160N B-tile (64 B/row) from W2 into LDS, hardware-padding
// each row by 16 B so the LDS stride is 80 B (B_STR). 2-D descriptor per
// CDNA5 ISA: data_size=4B, tile 16x160 DWORDs,
// tensor_dim0 = tensor_dim0_stride = K_TOT/2 DWORDs, pad 4 DW every 16 DW.
// ---------------------------------------------------------------------------
__device__ __forceinline__ void tdm_issue_b(const _Float16* gsrc, unsigned lds_off) {
  unsigned long long ga = (unsigned long long)(uintptr_t)gsrc;
  v4u g0;
  g0[0] = 1u;                                   // count=1, user descriptor
  g0[1] = lds_off;                              // LDS byte address
  g0[2] = (unsigned)ga;                         // global_addr[31:0]
  g0[3] = (unsigned)((ga >> 32) & 0x01FFFFFFu)  // global_addr[56:32]
          | 0x80000000u;                        // type=2 ("image")
  v8i g1;
  g1[0] = (int)((2u << 16)      // data_size = 4 bytes
              | (1u << 20)      // pad_enable
              | (3u << 22)      // pad_interval: 16 DWORDs
              | (3u << 25));    // pad_amount:   4 DWORDs
  g1[1] = (int)((unsigned)(K_TOT / 2) << 16);   // tensor_dim0[15:0]
  g1[2] = (int)((unsigned)NCOL << 16);          // tensor_dim1[15:0]
  g1[3] = (int)(16u << 16);                     // tile_dim0 = 16 DWORDs
  g1[4] = NCOL;                                 // tile_dim1 = 160 rows
  g1[5] = K_TOT / 2;                            // tensor_dim0_stride
  g1[6] = 0;
  g1[7] = 0;
  v4i z4 = {0, 0, 0, 0};
#if __clang_major__ >= 23
  v8i z8 = {0, 0, 0, 0, 0, 0, 0, 0};
  __builtin_amdgcn_tensor_load_to_lds(g0, g1, z4, z4, z8, 0);
#else
  __builtin_amdgcn_tensor_load_to_lds(g0, g1, z4, z4, 0);
#endif
}
#endif

// ---------------------------------------------------------------------------
// Fused KAN GEMM: D[64 x 160] += A[64 x 7072](features, on the fly)
//                               x B[7072 x 160](fused f16 weights), WMMA f32,
// B-tiles DMA'd by the Tensor Data Mover with LDS double buffering.
// ---------------------------------------------------------------------------
__global__ __launch_bounds__(BT) void kan_gemm(
    const float* __restrict__ X, const float* __restrict__ grid,
    const _Float16* __restrict__ W2,
    const float* __restrict__ w1, const float* __restrict__ b1,
    const float* __restrict__ w2, const float* __restrict__ b2,
    float* __restrict__ out)
{
  __shared__ __attribute__((aligned(16))) unsigned char smem[SMEM_BYTES];
  __shared__ float gsh[NGRID];

  _Float16* As   = (_Float16*)smem;                    // [MT * A_STR]
  _Float16* Bs   = (_Float16*)(smem + AS_BYTES);       // [2][NCOL * B_STR]
  float*    ybuf = (float*)smem;                       // epilogue reuse

  const int t    = threadIdx.x;
  const int wid  = t >> 5;
  const int lane = t & 31;
  const int half = lane >> 4;   // WMMA lane half
  const int lm   = lane & 15;   // M-row (A/C) or N-col (B/C) within fragment
  const int b0   = blockIdx.x * MT;

  if (t < NGRID) gsh[t] = grid[t];   // all 784 grid rows are identical

  v8f acc[HEADS];
#pragma unroll
  for (int nt = 0; nt < HEADS; ++nt)
#pragma unroll
    for (int i = 0; i < 8; ++i) acc[nt][i] = 0.0f;

  __syncthreads();

#if HAVE_TDM
  if (wid == 0) tdm_issue_b(W2, (unsigned)(uintptr_t)Bs);   // prefetch chunk 0
#endif

  for (int c = 0; c < CH_TOT; ++c) {
    const int buf = c & 1;
    _Float16* Bcur = Bs + buf * (NCOL * B_STR);

#if !HAVE_TDM
    // fallback: manual B-tile staging (u32 = packed f16 pair)
    {
      const uint32_t* src = (const uint32_t*)W2;
      uint32_t* dstB = (uint32_t*)Bcur;
      const int kb2 = c * 16;
#pragma unroll
      for (int u = t; u < NCOL * 16; u += BT) {
        int n = u >> 4, q = u & 15;
        dstB[n * (B_STR / 2) + q] = src[n * (K_TOT / 2) + kb2 + q];
      }
    }
#endif

    // ---- stage A-tile: 64 rows x 32 K of features (overlaps the TDM DMA) ----
    {
      const int r = t >> 1;                  // batch row within tile (0..63)
      if (c < CH_SP) {
        const int dbase = c * 4;
        const int dpair = (t & 1) * 2;
        const float* xp = X + (size_t)(b0 + r) * IN_DIM + dbase + dpair;
        float x0 = xp[0], x1 = xp[1];
        v8h f0 = spline8(x0, gsh);
        v8h f1 = spline8(x1, gsh);
        *(v8h*)(As + r * A_STR + dpair * 8)     = f0;
        *(v8h*)(As + r * A_STR + dpair * 8 + 8) = f1;
      } else {
        const int dbase = (c - CH_SP) * 32;
        const int h16   = (t & 1) * 16;
        v8h s0, s1;
        if (dbase + h16 < IN_DIM) {
          const float* xp = X + (size_t)(b0 + r) * IN_DIM + dbase + h16;
#pragma unroll
          for (int i = 0; i < 8; ++i) {
            s0[i] = (_Float16)silu(xp[i]);
            s1[i] = (_Float16)silu(xp[8 + i]);
          }
        } else {
#pragma unroll
          for (int i = 0; i < 8; ++i) { s0[i] = (_Float16)0.0f; s1[i] = (_Float16)0.0f; }
        }
        *(v8h*)(As + r * A_STR + h16)     = s0;
        *(v8h*)(As + r * A_STR + h16 + 8) = s1;
      }
    }

#if HAVE_TDM
    if (wid == 0) __builtin_amdgcn_s_wait_tensorcnt(0);   // B-tile c landed
#endif
    __syncthreads();   // As + Bcur visible; all waves done reading buf^1

#if HAVE_TDM
    if (wid == 0 && c + 1 < CH_TOT)                       // DMA next B-tile
      tdm_issue_b(W2 + (size_t)(c + 1) * 32,              // +64 B per chunk
                  (unsigned)(uintptr_t)(Bs + (buf ^ 1) * (NCOL * B_STR)));
#endif

    // ---- A fragment (16-bit A 16x32 layout): half0 K{0..7,16..23},
    //      half1 K{8..15,24..31}; row M = lane%16 ----
    const _Float16* arow = As + (wid * 16 + lm) * A_STR;
    v8h a0 = *(const v8h*)(arow + half * 8);
    v8h a1 = *(const v8h*)(arow + 16 + half * 8);
    v16h av;
#pragma unroll
    for (int i = 0; i < 8; ++i) { av[i] = a0[i]; av[i + 8] = a1[i]; }

    // ---- 10 N-tiles: B fragment (32x16): col N = lane%16, K = half*8+{0..7,16..23} ----
#pragma unroll
    for (int nt = 0; nt < HEADS; ++nt) {
      const _Float16* brow = Bcur + (nt * 16 + lm) * B_STR + half * 8;
      v8h bq0 = *(const v8h*)(brow);
      v8h bq1 = *(const v8h*)(brow + 16);
      v16h bv;
#pragma unroll
      for (int i = 0; i < 8; ++i) { bv[i] = bq0[i]; bv[i + 8] = bq1[i]; }
      acc[nt] = __builtin_amdgcn_wmma_f32_16x16x32_f16(
          false, av, false, bv, (short)0, acc[nt], false, false);
    }
    __syncthreads();   // protect As/Bs re-staging next iteration
  }

  // ---- epilogue: y = tanh(acc) -> LDS (C/D layout: M = i + 8*half) ----
  // (ybuf aliases the staging buffers; safe after the final barrier above)
#pragma unroll
  for (int nt = 0; nt < HEADS; ++nt)
#pragma unroll
    for (int i = 0; i < 8; ++i) {
      int rl = wid * 16 + half * 8 + i;
      ybuf[rl * YB_STR + nt * 16 + lm] = tanhf(acc[nt][i]);
    }
  __syncthreads();

  // ---- per-(row, head) MLP: 16 -> tanh(8) -> 1 ----
  for (int task = t; task < MT * HEADS; task += BT) {
    int h  = task % HEADS;
    int rr = task / HEADS;
    const float* yv = ybuf + rr * YB_STR + h * 16;
    float oa = b2[h];
#pragma unroll
    for (int p = 0; p < 8; ++p) {
      float s = b1[h * 8 + p];
      const float* w1p = w1 + (h * 8 + p) * 16;
#pragma unroll
      for (int o = 0; o < 16; ++o) s += yv[o] * w1p[o];
      oa += tanhf(s) * w2[h * 8 + p];
    }
    out[(size_t)(b0 + rr) * HEADS + h] = oa;
  }
}

// ---------------------------------------------------------------------------
extern "C" void kernel_launch(void* const* d_in, const int* in_sizes, int n_in,
                              void* d_out, int out_size, void* d_ws, size_t ws_size,
                              hipStream_t stream) {
  const float* X    = (const float*)d_in[0];   // (B,1,28,28) -> (B,784)
  const float* grid = (const float*)d_in[1];   // (784,12), rows identical
  const float* coef = (const float*)d_in[2];   // (10,784,16,8)
  const float* sb   = (const float*)d_in[3];   // (10,784,16)
  const float* ssp  = (const float*)d_in[4];   // (10,784,16)
  const float* mask = (const float*)d_in[5];   // (10,784,16)
  const float* w1   = (const float*)d_in[6];   // (10,8,16)
  const float* b1   = (const float*)d_in[7];   // (10,8)
  const float* w2   = (const float*)d_in[8];   // (10,1,8)
  const float* b2   = (const float*)d_in[9];   // (10,1)
  float* out = (float*)d_out;

  _Float16* W2 = (_Float16*)d_ws;              // 160*7072*2 = ~2.16 MB

  const int B = in_sizes[0] / IN_DIM;          // 16384

  const int totalW = NCOL * K_TOT;
  kan_prep_w<<<(totalW + 255) / 256, 256, 0, stream>>>(coef, sb, ssp, mask, W2);
  kan_gemm<<<B / MT, BT, 0, stream>>>(X, grid, W2, w1, b1, w2, b2, out);
}
